// BasicLayer_19653770347030
// MI455X (gfx1250) — compile-verified, matrix operand order checked by reading, not run on hardware
//
#include <hip/hip_runtime.h>
#include <math.h>

typedef __bf16 bf16;
typedef __attribute__((ext_vector_type(16))) __bf16 v16bf;
typedef __attribute__((ext_vector_type(8)))  float  v8f;

#define BATCH 2
#define CCH   96
#define C3    288
#define DD    32
#define HH    64
#define WW    64
#define NTOK  131072   // DD*HH*WW
#define HEADS 6
#define HC    16

// ---- workspace layout (bytes) ----
#define OFF_QKVD 0ull                                   // bf16 [B,288,N]   151 MB
#define OFF_QKV  150994944ull                           // bf16 [B,288,N]   151 MB (aliased by xres later)
#define OFF_XN   301989888ull                           // bf16 [B,N,96]    50 MB
#define OFF_S    352321536ull                           // f32  [B,6,16,16]
#define OFF_F    352333824ull                           // bf16 [B,96,96]
#define OFF_WBF  352370688ull                           // bf16 weights: qkv(27648)+fc1(18432)+fc2(18432)
#define OFF_XRES OFF_QKV                                // f32  [B,96,N]    100 MB (alias, qkv dead by then)

// ---------- WMMA fragment loaders (16x32 bf16 tile) ----------
// A: row-major MxK tile at p (ld elements per row). lane&15 = M row,
// VGPR j holds K pair at (j/4)*16 + (lane/16)*8 + (j%4)*2.
// B uses the mirrored layout (lane&15 = N column of B == row of B^T).
__device__ __forceinline__ v16bf load_frag(const bf16* __restrict__ p, size_t ld, int lane) {
    int m  = lane & 15;
    int hi = lane >> 4;
    const bf16* r = p + (size_t)m * ld;
    v16bf f;
#pragma unroll
    for (int j = 0; j < 8; ++j) {
        int kb = ((j >> 2) << 4) + (hi << 3) + ((j & 3) << 1);
        f[2 * j]     = r[kb];
        f[2 * j + 1] = r[kb + 1];
    }
    return f;
}

__device__ __forceinline__ v8f wmma_bf16(v16bf a, v16bf b, v8f c) {
    return __builtin_amdgcn_wmma_f32_16x16x32_bf16(false, a, false, b, (short)0, c, false, false);
}

__device__ __forceinline__ float gelu_exact(float x) {
    return 0.5f * x * (1.0f + erff(x * 0.70710678118654752f));
}

// ---------- K0: convert GEMM weights fp32 -> bf16 ----------
__global__ void k_wconv(const float* __restrict__ qkv_w, const float* __restrict__ fc1_w,
                        const float* __restrict__ fc2_w, bf16* __restrict__ wbf) {
    int i = blockIdx.x * 256 + threadIdx.x;            // 64512 total
    if (i < 27648)       wbf[i] = (bf16)qkv_w[i];
    else if (i < 46080)  wbf[i] = (bf16)fc1_w[i - 27648];
    else if (i < 64512)  wbf[i] = (bf16)fc2_w[i - 46080];
}

// ---------- K1: channel LayerNorm + transpose to token-major bf16 ----------
__global__ void k_ln1(const float* __restrict__ x, const float* __restrict__ w,
                      const float* __restrict__ b, bf16* __restrict__ xn) {
    __shared__ bf16 lds[256 * CCH];                    // 48 KB
    int blk  = blockIdx.x;                             // B * N/256 blocks
    int bb   = blk / (NTOK / 256);
    int nb   = (blk % (NTOK / 256)) * 256;
    int t    = threadIdx.x;
    const float* xb = x + (size_t)bb * CCH * NTOK + nb + t;
    float s = 0.f, s2 = 0.f;
#pragma unroll 4
    for (int c = 0; c < CCH; ++c) {
        float v = xb[(size_t)c * NTOK];
        s += v; s2 += v * v;
    }
    float mu   = s * (1.0f / CCH);
    float var  = s2 * (1.0f / CCH) - mu * mu;
    float rstd = rsqrtf(var + 1e-6f);
#pragma unroll 4
    for (int c = 0; c < CCH; ++c) {
        float v = xb[(size_t)c * NTOK];
        lds[t * CCH + c] = (bf16)(((v - mu) * rstd) * w[c] + b[c]);
    }
    __syncthreads();
    bf16* out = xn + ((size_t)bb * NTOK + nb) * CCH;
    for (int i = t; i < 256 * CCH; i += 256) out[i] = lds[i];
}

// ---------- K2: qkv = W(288x96) @ xn  (WMMA GEMM over token tiles) ----------
__global__ void k_qkv(const bf16* __restrict__ xn, const bf16* __restrict__ wbf,
                      const float* __restrict__ qkv_b, bf16* __restrict__ qkv) {
    int bb   = blockIdx.x >> 13;
    int nb   = (blockIdx.x & 8191) << 4;
    int lane = threadIdx.x & 31;
    int wid  = threadIdx.x >> 5;                       // 8 waves
    const bf16* xb = xn + ((size_t)bb * NTOK + nb) * CCH;
    v16bf bfrag[3];
#pragma unroll
    for (int f = 0; f < 3; ++f) bfrag[f] = load_frag(xb + f * 32, CCH, lane);
    for (int s = wid; s < 18; s += 8) {
        int oc0 = s * 16;
        v8f acc = {};
#pragma unroll
        for (int f = 0; f < 3; ++f) {
            v16bf a = load_frag(wbf + (size_t)oc0 * CCH + f * 32, CCH, lane);
            acc = wmma_bf16(a, bfrag[f], acc);
        }
        int hi = lane >> 4, col = lane & 15;
#pragma unroll
        for (int r = 0; r < 8; ++r) {
            int oc = oc0 + r + 8 * hi;
            qkv[((size_t)bb * C3 + oc) * NTOK + nb + col] = (bf16)(acc[r] + qkv_b[oc]);
        }
    }
}

// ---------- K3: depthwise 3x3x3 conv (SAME), bf16 in/out, L2-resident ----------
__global__ void k_dw(const bf16* __restrict__ qkv, const float* __restrict__ dw_w,
                     const float* __restrict__ dw_b, bf16* __restrict__ qkvd) {
    size_t id = (size_t)blockIdx.x * 256 + threadIdx.x; // B*288*N threads
    int n    = (int)(id & (NTOK - 1));
    int bc   = (int)(id >> 17);
    int ch   = bc % C3;
    int ww_  = n & 63;
    int hh_  = (n >> 6) & 63;
    int dd_  = n >> 12;
    const bf16* base = qkv + ((size_t)bc << 17);
    const float* wt  = dw_w + ch * 27;
    float sum = dw_b[ch];
#pragma unroll
    for (int dz = -1; dz <= 1; ++dz) {
        int z = dd_ + dz; if (z < 0 || z >= DD) continue;
#pragma unroll
        for (int dy = -1; dy <= 1; ++dy) {
            int y = hh_ + dy; if (y < 0 || y >= HH) continue;
#pragma unroll
            for (int dx = -1; dx <= 1; ++dx) {
                int xw = ww_ + dx; if (xw < 0 || xw >= WW) continue;
                int nn = (z << 12) + (y << 6) + xw;
                sum += (float)base[nn] * wt[(dz + 1) * 9 + (dy + 1) * 3 + (dx + 1)];
            }
        }
    }
    qkvd[id] = (bf16)sum;
}

// ---------- K4a: zero Gram accumulator ----------
__global__ void k_zero(float* __restrict__ S) {
    int i = blockIdx.x * 256 + threadIdx.x;
    if (i < BATCH * HEADS * 256) S[i] = 0.f;
}

// ---------- K4: S[b,h,c,d] += sum_n q[c,n]*k[d,n]  via bf16 WMMA ----------
__global__ void k_gram(const bf16* __restrict__ qkvd, float* __restrict__ S) {
    int bh    = blockIdx.x >> 6;                        // 12
    int chunk = blockIdx.x & 63;
    int bb = bh / HEADS, hd = bh % HEADS;
    int lane = threadIdx.x & 31, wid = threadIdx.x >> 5;
    const bf16* qp = qkvd + ((size_t)bb * C3 + hd * HC) * NTOK;
    const bf16* kp = qkvd + ((size_t)bb * C3 + CCH + hd * HC) * NTOK;
    int n0 = chunk * 2048 + wid * 256;
    v8f acc = {};
#pragma unroll
    for (int it = 0; it < 8; ++it) {
        int nn = n0 + it * 32;
        v16bf a = load_frag(qp + nn, NTOK, lane);
        v16bf b = load_frag(kp + nn, NTOK, lane);
        acc = wmma_bf16(a, b, acc);
    }
    int hi = lane >> 4, col = lane & 15;
#pragma unroll
    for (int r = 0; r < 8; ++r)
        atomicAdd(&S[bh * 256 + (r + 8 * hi) * 16 + col], acc[r]);
}

// ---------- K5: softmax with scale + mask ----------
__global__ void k_softmax(float* __restrict__ S, const float* __restrict__ mask) {
    int t = threadIdx.x;
    if (t >= BATCH * HEADS * HC) return;
    int bh = t >> 4, c = t & 15;
    int bb = bh / HEADS;
    const float scale = 0.40824829046386302f;           // 6^-0.5
    float a[16], mx = -1e30f;
#pragma unroll
    for (int d = 0; d < 16; ++d) {
        a[d] = S[bh * 256 + c * 16 + d] * scale + mask[bb * 256 + c * 16 + d];
        mx = fmaxf(mx, a[d]);
    }
    float sum = 0.f;
#pragma unroll
    for (int d = 0; d < 16; ++d) { a[d] = __expf(a[d] - mx); sum += a[d]; }
    float inv = 1.0f / sum;
#pragma unroll
    for (int d = 0; d < 16; ++d) S[bh * 256 + c * 16 + d] = a[d] * inv;
}

// ---------- K5b: F[b] = proj_w @ blockdiag(attn[b])  (96x96, bf16) ----------
__global__ void k_fuse(const float* __restrict__ S, const float* __restrict__ proj_w,
                       bf16* __restrict__ F) {
    int idx = blockIdx.x * 256 + threadIdx.x;           // B*96*96
    int bb  = idx / (CCH * CCH);
    int lin = idx % (CCH * CCH);
    int o = lin / CCH, i = lin % CCH;
    int hi = i >> 4, d = i & 15;
    float sum = 0.f;
#pragma unroll
    for (int c = 0; c < 16; ++c)
        sum += proj_w[o * CCH + hi * 16 + c] * S[(bb * HEADS + hi) * 256 + c * 16 + d];
    F[(size_t)bb * CCH * CCH + lin] = (bf16)sum;
}

// ---------- K6: xres = x + lam * (F @ v + proj_b)  (WMMA, LDS transpose) ----------
__global__ void k_attnout(const bf16* __restrict__ qkvd, const bf16* __restrict__ F,
                          const float* __restrict__ proj_b, const float* __restrict__ lam,
                          const float* __restrict__ x, float* __restrict__ xres) {
    __shared__ bf16 ldsV[16 * CCH];
    int bb = blockIdx.x >> 13;
    int nb = (blockIdx.x & 8191) << 4;
    int lane = threadIdx.x & 31, wid = threadIdx.x >> 5;  // 6 waves of 192
    const bf16* vp = qkvd + ((size_t)bb * C3 + 2 * CCH) * NTOK + nb;
    for (int idx = threadIdx.x; idx < 16 * CCH; idx += 192) {
        int ch = idx >> 4, t = idx & 15;
        ldsV[t * CCH + ch] = vp[(size_t)ch * NTOK + t];
    }
    __syncthreads();
    int oc0 = wid * 16;
    const bf16* Fb = F + (size_t)bb * CCH * CCH + (size_t)oc0 * CCH;
    v8f acc = {};
#pragma unroll
    for (int f = 0; f < 3; ++f) {
        v16bf a = load_frag(Fb + f * 32, CCH, lane);
        v16bf b = load_frag(ldsV + f * 32, CCH, lane);
        acc = wmma_bf16(a, b, acc);
    }
    int hi = lane >> 4, col = lane & 15;
#pragma unroll
    for (int r = 0; r < 8; ++r) {
        int oc = oc0 + r + 8 * hi;
        size_t gi = ((size_t)bb * CCH + oc) * NTOK + nb + col;
        xres[gi] = x[gi] + lam[oc] * (acc[r] + proj_b[oc]);
    }
}

// ---------- K7: out = fc2(gelu(fc1(ln2(xres)))) + xres  (fused WMMA MLP) ----------
__global__ void k_mlp(const float* __restrict__ xres, const float* __restrict__ ln2_w,
                      const float* __restrict__ ln2_b, const bf16* __restrict__ fc1_wb,
                      const float* __restrict__ fc1_b, const bf16* __restrict__ fc2_wb,
                      const float* __restrict__ fc2_b, float* __restrict__ out) {
    __shared__ float ldsX[16 * CCH];                    // residual tile (token-major)
    __shared__ bf16  ldsA[16 * CCH];                    // normalized activations
    __shared__ bf16  ldsT[16 * 192];                    // hidden (post-GELU)
    int bb = blockIdx.x >> 13;
    int nb = (blockIdx.x & 8191) << 4;
    int lane = threadIdx.x & 31, wid = threadIdx.x >> 5; // 8 waves of 256
    const float* xp = xres + (size_t)bb * CCH * NTOK + nb;
    for (int idx = threadIdx.x; idx < 16 * CCH; idx += 256) {
        int ch = idx >> 4, t = idx & 15;
        ldsX[t * CCH + ch] = xp[(size_t)ch * NTOK + t];
    }
    __syncthreads();
    if (threadIdx.x < 16) {
        int t = threadIdx.x;
        float s = 0.f, s2 = 0.f;
        for (int c = 0; c < CCH; ++c) { float v = ldsX[t * CCH + c]; s += v; s2 += v * v; }
        float mu = s * (1.0f / CCH);
        float rstd = rsqrtf(s2 * (1.0f / CCH) - mu * mu + 1e-6f);
        for (int c = 0; c < CCH; ++c) {
            float v = ldsX[t * CCH + c];
            ldsA[t * CCH + c] = (bf16)(((v - mu) * rstd) * ln2_w[c] + ln2_b[c]);
        }
    }
    __syncthreads();
    // fc1 (96 -> 192) + exact GELU, 12 strips over 8 waves
    for (int s = wid; s < 12; s += 8) {
        int oc0 = s * 16;
        v8f acc = {};
#pragma unroll
        for (int f = 0; f < 3; ++f) {
            v16bf a = load_frag(fc1_wb + (size_t)oc0 * CCH + f * 32, CCH, lane);
            v16bf b = load_frag(ldsA + f * 32, CCH, lane);
            acc = wmma_bf16(a, b, acc);
        }
        int hi = lane >> 4, col = lane & 15;
#pragma unroll
        for (int r = 0; r < 8; ++r) {
            int oc = oc0 + r + 8 * hi;
            ldsT[col * 192 + oc] = (bf16)gelu_exact(acc[r] + fc1_b[oc]);
        }
    }
    __syncthreads();
    // fc2 (192 -> 96) + residual, 6 strips
    if (wid < 6) {
        int oc0 = wid * 16;
        v8f acc = {};
#pragma unroll
        for (int f = 0; f < 6; ++f) {
            v16bf a = load_frag(fc2_wb + (size_t)oc0 * 192 + f * 32, 192, lane);
            v16bf b = load_frag(ldsT + f * 32, 192, lane);
            acc = wmma_bf16(a, b, acc);
        }
        int hi = lane >> 4, col = lane & 15;
#pragma unroll
        for (int r = 0; r < 8; ++r) {
            int oc = oc0 + r + 8 * hi;
            out[((size_t)bb * CCH + oc) * NTOK + nb + col] =
                acc[r] + fc2_b[oc] + ldsX[col * CCH + oc];
        }
    }
}

extern "C" void kernel_launch(void* const* d_in, const int* in_sizes, int n_in,
                              void* d_out, int out_size, void* d_ws, size_t ws_size,
                              hipStream_t stream) {
    const float* x      = (const float*)d_in[0];
    const float* mask   = (const float*)d_in[1];
    const float* ln1_w  = (const float*)d_in[2];
    const float* ln1_b  = (const float*)d_in[3];
    const float* qkv_w  = (const float*)d_in[4];
    const float* qkv_b  = (const float*)d_in[5];
    const float* dw_w   = (const float*)d_in[6];
    const float* dw_b   = (const float*)d_in[7];
    const float* proj_w = (const float*)d_in[8];
    const float* proj_b = (const float*)d_in[9];
    const float* lam    = (const float*)d_in[10];
    const float* ln2_w  = (const float*)d_in[11];
    const float* ln2_b  = (const float*)d_in[12];
    const float* fc1_b  = (const float*)d_in[14];
    const float* fc2_b  = (const float*)d_in[16];
    const float* fc1_w  = (const float*)d_in[13];
    const float* fc2_w  = (const float*)d_in[15];

    char* ws = (char*)d_ws;
    bf16*  qkvd = (bf16*)(ws + OFF_QKVD);
    bf16*  qkv  = (bf16*)(ws + OFF_QKV);
    bf16*  xn   = (bf16*)(ws + OFF_XN);
    float* S    = (float*)(ws + OFF_S);
    bf16*  F    = (bf16*)(ws + OFF_F);
    bf16*  wbf  = (bf16*)(ws + OFF_WBF);
    float* xres = (float*)(ws + OFF_XRES);              // alias of qkv (dead after k_dw)
    float* out  = (float*)d_out;

    int tilesPerB = NTOK / 16;                          // 8192

    k_wconv  <<<252, 256, 0, stream>>>(qkv_w, fc1_w, fc2_w, wbf);
    k_ln1    <<<BATCH * (NTOK / 256), 256, 0, stream>>>(x, ln1_w, ln1_b, xn);
    k_qkv    <<<BATCH * tilesPerB, 256, 0, stream>>>(xn, wbf, qkv_b, qkv);
    k_dw     <<<(BATCH * C3 * NTOK) / 256, 256, 0, stream>>>(qkv, dw_w, dw_b, qkvd);
    k_zero   <<<12, 256, 0, stream>>>(S);
    k_gram   <<<BATCH * HEADS * 64, 256, 0, stream>>>(qkvd, S);
    k_softmax<<<1, 256, 0, stream>>>(S, mask);
    k_fuse   <<<(BATCH * CCH * CCH) / 256, 256, 0, stream>>>(S, proj_w, F);
    k_attnout<<<BATCH * tilesPerB, 192, 0, stream>>>(qkvd, F, proj_b, lam, x, xres);
    k_mlp    <<<BATCH * tilesPerB, 256, 0, stream>>>(xres, ln2_w, ln2_b,
                                                     wbf + 27648, fc1_b,
                                                     wbf + 46080, fc2_b, out);
}
